// Encoder_50208167690380
// MI455X (gfx1250) — matrix operand order repaired and checked by reading, not compiled, not run on hardware
//
#include <hip/hip_runtime.h>
#include <hip/hip_bf16.h>

// ---------------------------------------------------------------------------
// LSTM encoder for MI455X (gfx1250), wave32 + v_wmma_f32_16x16x32_bf16,
// with double-buffered GLOBAL_LOAD_ASYNC_TO_LDS staging in the recurrent GEMM.
// Dims: B=64, S=512, V=32000, E=512, H=1024, 4H=4096.
// ---------------------------------------------------------------------------

typedef __attribute__((ext_vector_type(16))) __bf16 v16bf;
typedef __attribute__((ext_vector_type(8)))  __bf16 v8bf;
typedef __attribute__((ext_vector_type(8)))  float  v8f;

#define B_DIM   64
#define S_DIM   512
#define E_DIM   512
#define H_DIM   1024
#define G_DIM   4096   // 4*H
#define M_BIG   (S_DIM * B_DIM)   // 32768

// ---- CDNA5 async copy (global -> LDS) --------------------------------------
// Builtin signature (from clang diagnostic): first param is
//   'int __attribute__((vector_size(16))) __device__ *'  (AS1, b128 payload)
typedef int b128_t __attribute__((vector_size(16)));
typedef __attribute__((address_space(1))) b128_t* gp128_t;
typedef __attribute__((address_space(3))) b128_t* lp128_t;

#if defined(__has_builtin) && __has_builtin(__builtin_amdgcn_global_load_async_to_lds_b128)
#define ASYNC_CP_B128(g, l)                                                    \
    __builtin_amdgcn_global_load_async_to_lds_b128((gp128_t)(g), (lp128_t)(l), 0, 0)
#else
#define ASYNC_CP_B128(g, l)                                                    \
    do {                                                                       \
        unsigned _lo = (unsigned)(size_t)(lp128_t)(l);                         \
        asm volatile("global_load_async_to_lds_b128 %0, %1, off"               \
                     :: "v"(_lo), "v"(g) : "memory");                          \
    } while (0)
#endif

#if defined(__has_builtin) && __has_builtin(__builtin_amdgcn_s_wait_asynccnt)
#define ASYNC_WAIT() __builtin_amdgcn_s_wait_asynccnt(0)
#else
#define ASYNC_WAIT() asm volatile("s_wait_asynccnt 0x0" ::: "memory")
#endif

// ---- float -> bf16 with round-to-nearest-even ------------------------------
__device__ __forceinline__ __bf16 f2bf(float f) {
    unsigned x = __float_as_uint(f);
    unsigned r = (x + 0x7FFFu + ((x >> 16) & 1u)) >> 16;
    unsigned short us = (unsigned short)r;
    __bf16 out;
    __builtin_memcpy(&out, &us, 2);
    return out;
}

__device__ __forceinline__ float sigf(float x) {
    return 1.0f / (1.0f + __expf(-x));
}

// ---- WMMA fragment loads (ISA 7.12.2 layouts, wave32) ----------------------
// A (16x32 bf16, row-major src [M,K]): lane m = lane%16;
//   lanes 0-15 : halves 0-7 -> k0..k7,   halves 8-15 -> k16..k23
//   lanes 16-31: halves 0-7 -> k8..k15,  halves 8-15 -> k24..k31
__device__ __forceinline__ v16bf load_a_frag(const __bf16* __restrict__ A,
                                             int lda, int row0, int k0, int lane) {
    const int m  = lane & 15;
    const int kb = (lane & 16) ? 8 : 0;
    const __bf16* p = A + (size_t)(row0 + m) * lda + (k0 + kb);
    v8bf lo = *(const v8bf*)(p);        // k = kb .. kb+7
    v8bf hi = *(const v8bf*)(p + 16);   // k = 16+kb .. 16+kb+7
    return __builtin_shufflevector(lo, hi, 0,1,2,3,4,5,6,7,8,9,10,11,12,13,14,15);
}

// Same A layout, but from an LDS-resident 64x32 panel (row stride 32).
__device__ __forceinline__ v16bf load_a_frag_lds(const __bf16* As, int row0, int lane) {
    const int m  = row0 + (lane & 15);
    const int kb = (lane & 16) ? 8 : 0;
    const __bf16* p = As + m * 32 + kb;
    v8bf lo = *(const v8bf*)(p);
    v8bf hi = *(const v8bf*)(p + 16);
    return __builtin_shufflevector(lo, hi, 0,1,2,3,4,5,6,7,8,9,10,11,12,13,14,15);
}

// B (32x16 bf16, "column-major" src: Wt[N][K] row-major): lane n = lane%16;
//   lanes 0-15 hold k 0..7, 16..23; lanes 16-31 hold k 8..15, 24..31.
__device__ __forceinline__ v16bf load_b_frag(const __bf16* __restrict__ Wt,
                                             int ldk, int col0, int k0, int lane) {
    const int n  = col0 + (lane & 15);
    const int kb = (lane & 16) ? 8 : 0;
    const __bf16* p = Wt + (size_t)n * ldk + (k0 + kb);
    v8bf lo = *(const v8bf*)(p);
    v8bf hi = *(const v8bf*)(p + 16);
    return __builtin_shufflevector(lo, hi, 0,1,2,3,4,5,6,7,8,9,10,11,12,13,14,15);
}

__device__ __forceinline__ v8f wmma_bf16(v16bf a, v16bf b, v8f c) {
    return __builtin_amdgcn_wmma_f32_16x16x32_bf16(
        /*neg_a=*/false, a, /*neg_b=*/false, b,
        /*c_mod=*/(short)0, c, /*reuse_a=*/false, /*reuse_b=*/false);
}

// C/D (16x16 f32): VGPR r, lanes 0-15 -> M=r, lanes 16-31 -> M=8+r; N=lane%16.
__device__ __forceinline__ void store_tile_addsrc(const v8f& acc,
                                                  float* __restrict__ C, int ldc,
                                                  int row0, int col0,
                                                  const float* __restrict__ addsrc,
                                                  int ldadd, int lane) {
    const int n  = col0 + (lane & 15);
    const int mb = (lane & 16) ? 8 : 0;
#pragma unroll
    for (int r = 0; r < 8; ++r) {
        const int m = row0 + mb + r;
        C[(size_t)m * ldc + n] = acc[r] + addsrc[(size_t)m * ldadd + n];
    }
}

__device__ __forceinline__ void store_tile_bias(const v8f& acc,
                                                float* __restrict__ C, int ldc,
                                                int row0, int col0,
                                                const float* __restrict__ bias,
                                                int lane) {
    const int n  = col0 + (lane & 15);
    const int mb = (lane & 16) ? 8 : 0;
    const float bv = bias[n];
#pragma unroll
    for (int r = 0; r < 8; ++r) {
        const int m = row0 + mb + r;
        C[(size_t)m * ldc + n] = acc[r] + bv;
    }
}

// ---------------------------------------------------------------------------
// Prep kernels
// ---------------------------------------------------------------------------

// W [K][N] fp32 -> Wt [N][K] bf16   (one-time, ~25 MB traffic)
__global__ void transpose_convert_w(const float* __restrict__ W,
                                    __bf16* __restrict__ Wt, int K, int N) {
    size_t idx = (size_t)blockIdx.x * blockDim.x + threadIdx.x;  // over K*N
    int k = (int)(idx / N);
    int n = (int)(idx % N);
    Wt[(size_t)n * K + k] = f2bf(W[idx]);
}

// X[s*B + b][e] = bf16(embedding[seq[b][s]][e]); float4 per thread.
__global__ void gather_embed(const int* __restrict__ seq,
                             const float* __restrict__ emb,
                             __bf16* __restrict__ X) {
    size_t t = (size_t)blockIdx.x * blockDim.x + threadIdx.x;  // M_BIG*E/4 threads
    int row = (int)(t >> 7);              // E/4 = 128 vec4 per row
    int e4  = (int)(t & 127) << 2;
    int s = row >> 6;                     // row = s*64 + b
    int b = row & 63;
    int tok = seq[(size_t)b * S_DIM + s];
    const float4 v = *(const float4*)(emb + (size_t)tok * E_DIM + e4);
    __bf16* o = X + (size_t)row * E_DIM + e4;
    o[0] = f2bf(v.x); o[1] = f2bf(v.y); o[2] = f2bf(v.z); o[3] = f2bf(v.w);
}

__global__ void init_state(float* __restrict__ h, float* __restrict__ c,
                           __bf16* __restrict__ hbf) {
    int tid = blockIdx.x * blockDim.x + threadIdx.x;   // 65536
    h[tid] = 0.0f;
    c[tid] = 0.0f;
    hbf[tid] = f2bf(0.0f);
}

// ---------------------------------------------------------------------------
// Big input-projection GEMM: xproj[M_BIG,4096] = X[M_BIG,512] @ WihT^T + bias
// Block = 256 threads = 8 waves arranged 2x4; block tile 64 x 256;
// each wave computes a 32x64 region as 2x4 WMMA tiles: 8 WMMA per 6 frag loads.
// Grid: (4096/256, 32768/64) = (16, 512). Exact — EXEC all 1s everywhere.
// ---------------------------------------------------------------------------
__global__ void __launch_bounds__(256)
xproj_gemm(const __bf16* __restrict__ X, const __bf16* __restrict__ WihT,
           const float* __restrict__ bias, float* __restrict__ xproj) {
    const int lane = threadIdx.x & 31;
    const int wave = threadIdx.x >> 5;
    const int Row0 = blockIdx.y * 64 + (wave >> 2) * 32;   // wave m: 0..1
    const int Col0 = blockIdx.x * 256 + (wave & 3) * 64;   // wave n: 0..3

    v8f acc[2][4] = {};
#pragma unroll 2
    for (int k0 = 0; k0 < E_DIM; k0 += 32) {
        v16bf a0 = load_a_frag(X, E_DIM, Row0,      k0, lane);
        v16bf a1 = load_a_frag(X, E_DIM, Row0 + 16, k0, lane);
#pragma unroll
        for (int j = 0; j < 4; ++j) {
            v16bf b = load_b_frag(WihT, E_DIM, Col0 + 16 * j, k0, lane);
            acc[0][j] = wmma_bf16(a0, b, acc[0][j]);
            acc[1][j] = wmma_bf16(a1, b, acc[1][j]);
        }
    }
#pragma unroll
    for (int i = 0; i < 2; ++i)
#pragma unroll
        for (int j = 0; j < 4; ++j)
            store_tile_bias(acc[i][j], xproj, G_DIM,
                            Row0 + 16 * i, Col0 + 16 * j, bias, lane);
}

// ---------------------------------------------------------------------------
// Recurrent step GEMM: gates[64,4096] = hbf[64,1024] @ WhhT^T + xproj_s
// Block = 8 waves as (4 m-tiles x 2 n-tiles) -> block tile 64 x 32.
// Grid = 4096/32 = 128 blocks exactly.
// A panel (64 x 32 bf16 per K-step, 4 KB) is double-buffered in LDS via
// GLOBAL_LOAD_ASYNC_TO_LDS_B128 (ASYNCcnt), WMMA A-frags come from ds_load.
// B (W_hh bf16, 8.4 MB) streams from global and stays L2-resident (192 MB L2).
// ---------------------------------------------------------------------------
__global__ void __launch_bounds__(256)
step_gemm(const __bf16* __restrict__ hbf, const __bf16* __restrict__ WhhT,
          const float* __restrict__ xproj_s, float* __restrict__ gates) {
    __shared__ __align__(16) __bf16 smA[2][B_DIM * 32];   // 2 x 4 KB

    const int tid  = threadIdx.x;
    const int lane = tid & 31;
    const int wave = tid >> 5;
    const int Row0 = (wave & 3) * 16;                      // 0..48
    const int Col0 = blockIdx.x * 32 + (wave >> 2) * 16;   // n tile

    // async staging: 256 threads x 16 B cover the 64x32 bf16 panel
    const int arow = tid >> 2;            // 0..63
    const int acol = (tid & 3) * 8;       // 0,8,16,24 (bf16 elems)

    {   // prologue: stage k-panel 0 into buffer 0
        const __bf16* g = hbf + (size_t)arow * H_DIM + acol;
        ASYNC_CP_B128(g, &smA[0][arow * 32 + acol]);
    }

    v8f acc = {};
    for (int kk = 0; kk < H_DIM / 32; ++kk) {
        ASYNC_WAIT();            // panel kk resident in LDS
        __syncthreads();
        if (kk + 1 < H_DIM / 32) {
            const __bf16* g = hbf + (size_t)arow * H_DIM + (kk + 1) * 32 + acol;
            ASYNC_CP_B128(g, &smA[(kk + 1) & 1][arow * 32 + acol]);
        }
        v16bf a = load_a_frag_lds(smA[kk & 1], Row0, lane);
        v16bf b = load_b_frag(WhhT, H_DIM, Col0, kk * 32, lane);
        acc = wmma_bf16(a, b, acc);
    }
    store_tile_addsrc(acc, gates, G_DIM, Row0, Col0, xproj_s, G_DIM, lane);
}

// ---------------------------------------------------------------------------
// Elementwise LSTM cell update. h, c live directly in d_out.
// ---------------------------------------------------------------------------
__global__ void lstm_update(const float* __restrict__ gates,
                            float* __restrict__ h, float* __restrict__ c,
                            __bf16* __restrict__ hbf) {
    int tid = blockIdx.x * blockDim.x + threadIdx.x;   // 65536
    int b = tid >> 10;
    int j = tid & 1023;
    const float* g = gates + (size_t)b * G_DIM;
    float ig = sigf(g[j]);
    float fg = sigf(g[H_DIM + j]);
    float gg = tanhf(g[2 * H_DIM + j]);
    float og = sigf(g[3 * H_DIM + j]);
    float cn = fg * c[tid] + ig * gg;
    float hn = og * tanhf(cn);
    c[tid]   = cn;
    h[tid]   = hn;
    hbf[tid] = f2bf(hn);
}

// ---------------------------------------------------------------------------
extern "C" void kernel_launch(void* const* d_in, const int* in_sizes, int n_in,
                              void* d_out, int out_size, void* d_ws, size_t ws_size,
                              hipStream_t stream) {
    const int*   seq  = (const int*)d_in[0];     // [B,S]
    const float* emb  = (const float*)d_in[1];   // [V,E]
    const float* Wih  = (const float*)d_in[2];   // [E,4H]
    const float* Whh  = (const float*)d_in[3];   // [H,4H]
    const float* bias = (const float*)d_in[4];   // [4H]

    float* h = (float*)d_out;                    // [64,1024]
    float* c = h + B_DIM * H_DIM;                // [64,1024]

    // ---- workspace layout (all regions 16B aligned) ----
    __bf16* WihT = (__bf16*)d_ws;                              // [4096,512]
    __bf16* WhhT = WihT + (size_t)G_DIM * E_DIM;               // [4096,1024]
    __bf16* X    = WhhT + (size_t)G_DIM * H_DIM;               // [32768,512]
    __bf16* hbf  = X + (size_t)M_BIG * E_DIM;                  // [64,1024]
    float*  xproj = (float*)(hbf + B_DIM * H_DIM);             // [32768,4096]
    float*  gates = xproj + (size_t)M_BIG * G_DIM;             // [64,4096]
    (void)ws_size; (void)in_sizes; (void)n_in; (void)out_size;

    // ---- one-time prep ----
    transpose_convert_w<<<(E_DIM * G_DIM) / 256, 256, 0, stream>>>(Wih, WihT, E_DIM, G_DIM);
    transpose_convert_w<<<(H_DIM * G_DIM) / 256, 256, 0, stream>>>(Whh, WhhT, H_DIM, G_DIM);
    gather_embed<<<(M_BIG * (E_DIM / 4)) / 256, 256, 0, stream>>>(seq, emb, X);
    init_state<<<(B_DIM * H_DIM) / 256, 256, 0, stream>>>(h, c, hbf);

    // ---- big input projection (parallel, WMMA) ----
    dim3 gridBig(G_DIM / 256, M_BIG / 64);
    xproj_gemm<<<gridBig, 256, 0, stream>>>(X, WihT, bias, xproj);

    // ---- sequential scan: 512 steps of (WMMA step-GEMM, cell update) ----
    for (int s = 0; s < S_DIM; ++s) {
        const float* xs = xproj + (size_t)s * B_DIM * G_DIM;
        step_gemm<<<128, 256, 0, stream>>>(hbf, WhhT, xs, gates);
        lstm_update<<<(B_DIM * H_DIM) / 256, 256, 0, stream>>>(gates, h, c, hbf);
    }
}
